// im_encoder_60876866453802
// MI455X (gfx1250) — compile-verified, matrix operand order checked by reading, not run on hardware
//
#include <hip/hip_runtime.h>

#define KNN   16
#define NPTS  2048
#define BSIZE 8
#define DDIM  3
#define MROWS (NPTS * KNN)   /* 32768 rows per batch sample */
#define EPSV  1e-5f

typedef __attribute__((ext_vector_type(16))) __bf16 v16bf;
typedef __attribute__((ext_vector_type(8)))  __bf16 v8bf;
typedef __attribute__((ext_vector_type(8)))  float  v8f;

// Build a 16-element bf16 WMMA fragment from two 16-byte LDS chunks.
__device__ __forceinline__ v16bf ldfrag(const __bf16* p) {
    v8bf lo = *(const v8bf*)(p);
    v8bf hi = *(const v8bf*)(p + 16);
    return __builtin_shufflevector(lo, hi,
        0, 1, 2, 3, 4, 5, 6, 7, 8, 9, 10, 11, 12, 13, 14, 15);
}

// Byte offset of an LDS pointer within the workgroup's LDS allocation
// (ptrtoint of an addrspace(3) pointer == LDS offset, which is what the
// async-to-LDS instructions take in their VDST operand).
__device__ __forceinline__ unsigned lds_off(const void* p) {
    return (unsigned)(unsigned long long)(__attribute__((address_space(3))) const char*)p;
}

// Async DMA of 16 bytes global -> LDS (per lane), tracked by ASYNCcnt.
__device__ __forceinline__ void async_copy_b128(unsigned ldsDst, const void* gsrc) {
    asm volatile("global_load_async_to_lds_b128 %0, %1, off"
                 :: "v"(ldsDst), "v"(gsrc) : "memory");
}
__device__ __forceinline__ void wait_async0() {
    asm volatile("s_wait_asynccnt 0x0" ::: "memory");
}

// ---------------------------------------------------------------------------
// KNN: one thread per (batch, point). Sorted insertion of 16 smallest d^2,
// strict compares reproduce top_k stability (earlier index wins ties).
// ---------------------------------------------------------------------------
__global__ void knn_kernel(const float* __restrict__ x, int* __restrict__ idx) {
    int gid = blockIdx.x * blockDim.x + threadIdx.x;
    if (gid >= BSIZE * NPTS) return;
    int b = gid / NPTS, i = gid % NPTS;
    const float* base = x + (size_t)b * NPTS * DDIM;
    float xi = base[i * 3 + 0], yi = base[i * 3 + 1], zi = base[i * 3 + 2];
    float best[KNN];
    int   bidx[KNN];
#pragma unroll
    for (int k = 0; k < KNN; k++) { best[k] = 3.0e38f; bidx[k] = 0; }
    for (int j = 0; j < NPTS; j++) {
        if (j == i) continue;                       // diag gets +1e10 in ref
        float dx = base[j * 3 + 0] - xi;
        float dy = base[j * 3 + 1] - yi;
        float dz = base[j * 3 + 2] - zi;
        float d = dx * dx + dy * dy + dz * dz;
        if (d < best[KNN - 1]) {
            int p = KNN - 1;
            while (p > 0 && best[p - 1] > d) {
                best[p] = best[p - 1]; bidx[p] = bidx[p - 1]; --p;
            }
            best[p] = d; bidx[p] = j;
        }
    }
    for (int k = 0; k < KNN; k++) idx[(size_t)gid * KNN + k] = bidx[k];
}

// ---------------------------------------------------------------------------
// Build edge features t = [h_i, h_j - h_i], zero-padded to Kd cols, bf16.
// ---------------------------------------------------------------------------
__global__ void build_t_kernel(const float* __restrict__ h, const int* __restrict__ idx,
                               __bf16* __restrict__ T, int cin, int Kd) {
    size_t gid = (size_t)blockIdx.x * blockDim.x + threadIdx.x;
    size_t total = (size_t)MROWS * Kd;
    if (gid >= total) return;
    int c = (int)(gid % Kd);
    int m = (int)(gid / Kd);
    int i = m / KNN, k = m % KNN;
    float v = 0.0f;
    if (c < cin) {
        v = h[(size_t)i * cin + c];
    } else if (c < 2 * cin) {
        int j  = idx[(size_t)i * KNN + k];
        int cc = c - cin;
        v = h[(size_t)j * cin + cc] - h[(size_t)i * cin + cc];
    }
    T[gid] = (__bf16)v;
}

// Convert fp32 weight (Krows x Nc) to bf16, zero-padded to (Kd x Nc).
__global__ void convert_w_kernel(const float* __restrict__ w, __bf16* __restrict__ Wb,
                                 int Krows, int Kd, int Nc) {
    int gid = blockIdx.x * blockDim.x + threadIdx.x;
    if (gid >= Kd * Nc) return;
    int r = gid / Nc, c = gid % Nc;
    Wb[gid] = (__bf16)((r < Krows) ? w[(size_t)r * Nc + c] : 0.0f);
}

// ---------------------------------------------------------------------------
// WMMA GEMM: C(M x NC) = A(M x KD) @ B(KD x NC) + bias, bf16 in, f32 acc,
// bf16 out. Compile-time KD/NC -> immediate-offset addressing in hot loop.
// Block = 256 threads (8 wave32), block tile 128x64, K-step 32.
// Waves in 4x2 grid; each wave owns a 32x32 output tile -> 4 WMMAs/K-step.
// A tile staged with GLOBAL_LOAD_ASYNC_TO_LDS_B128 (ASYNCcnt), B tile staged
// through VGPRs (LDS-transposed). Double-buffered LDS, one barrier per step.
// LDS rows padded to 40 bf16 (20 dwords) -> conflict-free 16-lane reads.
// ---------------------------------------------------------------------------
template <int KD, int NC>
__global__ __launch_bounds__(256)
void gemm_wmma_kernel(const __bf16* __restrict__ A, const __bf16* __restrict__ B,
                      const float* __restrict__ bias, __bf16* __restrict__ C) {
    __shared__ __bf16 As[2][128 * 40];   // 128 rows x 32 K, stride 40
    __shared__ __bf16 Bs[2][64 * 40];    // 64 cols x 32 K, transposed, stride 40

    const int tid    = threadIdx.x;
    const int wave   = tid >> 5;
    const int lane   = tid & 31;
    const int laneHi = lane >> 4;        // 0 or 1
    const int ln     = lane & 15;
    const int waveM  = wave & 3;         // 4 row groups of 32
    const int waveN  = wave >> 2;        // 2 col groups of 32
    const int rowTile = blockIdx.y * 128;
    const int colTile = blockIdx.x * 64;

    // Staging assignments (uniform across K-steps).
    const int ar   = tid >> 1;           // A row 0..127
    const int ah8  = (tid & 1) << 4;     // 0 or 16 within the 32-wide K slab
    const int bcol = tid & 63;           // B column
    const int bkb  = (tid >> 6) << 3;    // B k-base (0,8,16,24)

    const __bf16* aSrcBase = A + (size_t)(rowTile + ar) * KD + ah8;
    const __bf16* bSrcBase = B + (size_t)bkb * NC + colTile + bcol;
    const unsigned aDst0 = lds_off(&As[0][ar * 40 + ah8]);
    const unsigned aDst1 = lds_off(&As[1][ar * 40 + ah8]);

    v8f acc[2][2];
#pragma unroll
    for (int xi = 0; xi < 2; xi++)
#pragma unroll
        for (int yi = 0; yi < 2; yi++)
#pragma unroll
            for (int e = 0; e < 8; e++) acc[xi][yi][e] = 0.0f;

    __bf16 bR[8];                        // per-thread staged B (16 B, strided)

    auto asyncA = [&](int kk, int buf) { // 32 B/thread, two async b128 DMAs
        const __bf16* src = aSrcBase + kk;
        unsigned dst = buf ? aDst1 : aDst0;
        async_copy_b128(dst,      src);
        async_copy_b128(dst + 16, src + 8);
    };
    auto loadB = [&](int kk) {
        const __bf16* bs = bSrcBase + (size_t)kk * NC;
#pragma unroll
        for (int q = 0; q < 8; q++) bR[q] = bs[(size_t)q * NC];
    };
    auto storeB = [&](int buf) {
        __bf16* bd = &Bs[buf][bcol * 40 + bkb];
#pragma unroll
        for (int q = 0; q < 8; q++) bd[q] = bR[q];
    };

    asyncA(0, 0);
    loadB(0);
    storeB(0);
    wait_async0();
    __syncthreads();

    int cur = 0;
#pragma unroll
    for (int kk = 0; kk < KD; kk += 32) {
        const bool hasNext = (kk + 32) < KD;
        if (hasNext) {
            asyncA(kk + 32, cur ^ 1);
            loadB(kk + 32);
        }

        // A frags: lanes 0-15 -> K {0..7,16..23}; lanes 16-31 -> K {8..15,24..31}
        const __bf16* a0p = &As[cur][(waveM * 32 + ln) * 40 + laneHi * 8];
        v16bf a0 = ldfrag(a0p);
        v16bf a1 = ldfrag(a0p + 16 * 40);
        const __bf16* b0p = &Bs[cur][(waveN * 32 + ln) * 40 + laneHi * 8];
        v16bf b0 = ldfrag(b0p);
        v16bf b1 = ldfrag(b0p + 16 * 40);

        acc[0][0] = __builtin_amdgcn_wmma_f32_16x16x32_bf16(
            false, a0, false, b0, (short)0, acc[0][0], false, false);
        acc[0][1] = __builtin_amdgcn_wmma_f32_16x16x32_bf16(
            false, a0, false, b1, (short)0, acc[0][1], false, false);
        acc[1][0] = __builtin_amdgcn_wmma_f32_16x16x32_bf16(
            false, a1, false, b0, (short)0, acc[1][0], false, false);
        acc[1][1] = __builtin_amdgcn_wmma_f32_16x16x32_bf16(
            false, a1, false, b1, (short)0, acc[1][1], false, false);

        if (hasNext) {
            storeB(cur ^ 1);
            wait_async0();     // next A tile fully landed in LDS
            __syncthreads();   // orders reads(cur) vs next writes(cur), writes(cur^1) vs reads
        }
        cur ^= 1;
    }

    // Writeback per ISA C/D layout: VGPR r -> M = r + 8*laneHi, N = lane&15.
    const int orow0 = rowTile + waveM * 32;
    const int ocol0 = colTile + waveN * 32;
#pragma unroll
    for (int xi = 0; xi < 2; xi++) {
        const int rbase = orow0 + xi * 16 + laneHi * 8;
#pragma unroll
        for (int yi = 0; yi < 2; yi++) {
            const int ocol = ocol0 + yi * 16 + ln;
            const float bv = bias[ocol];
#pragma unroll
            for (int r = 0; r < 8; r++) {
                float v = acc[xi][yi][r] + bv;
                C[(size_t)(rbase + r) * NC + ocol] = (__bf16)v;
            }
        }
    }
}

// ---------------------------------------------------------------------------
// Column sums / sums-of-squares over M rows (one block per column).
// Deterministic fixed-order reduction (BN stats must be bit-stable).
// ---------------------------------------------------------------------------
__global__ void colstats_kernel(const __bf16* __restrict__ Xm, int M, int Nc,
                                float* __restrict__ sum, float* __restrict__ sumsq) {
    int col = blockIdx.x;
    float s = 0.0f, s2 = 0.0f;
    for (int r = threadIdx.x; r < M; r += blockDim.x) {
        float v = (float)Xm[(size_t)r * Nc + col];
        s += v; s2 += v * v;
    }
    __shared__ float sh[256], sh2[256];
    sh[threadIdx.x] = s; sh2[threadIdx.x] = s2;
    __syncthreads();
    for (int o = 128; o > 0; o >>= 1) {
        if (threadIdx.x < o) {
            sh[threadIdx.x]  += sh[threadIdx.x + o];
            sh2[threadIdx.x] += sh2[threadIdx.x + o];
        }
        __syncthreads();
    }
    if (threadIdx.x == 0) { sum[col] = sh[0]; sumsq[col] = sh2[0]; }
}

__global__ void bnfinalize_kernel(const float* __restrict__ sum, const float* __restrict__ sumsq,
                                  const float* __restrict__ g, const float* __restrict__ be,
                                  int M, float* __restrict__ scale, float* __restrict__ shift) {
    int c = threadIdx.x;
    float inv  = 1.0f / (float)M;
    float mean = sum[c] * inv;
    float var  = sumsq[c] * inv - mean * mean;
    float sc   = g[c] * rsqrtf(var + EPSV);
    scale[c] = sc;
    shift[c] = be[c] - mean * sc;
}

// BN + ReLU elementwise -> bf16 (feeds the second GEMM).
__global__ void bnrelu_kernel(const __bf16* __restrict__ X, const float* __restrict__ scale,
                              const float* __restrict__ shift, __bf16* __restrict__ Y,
                              int M, int Nc) {
    size_t gid = (size_t)blockIdx.x * blockDim.x + threadIdx.x;
    if (gid >= (size_t)M * Nc) return;
    int c = (int)(gid % Nc);
    float v = (float)X[gid] * scale[c] + shift[c];
    Y[gid] = (__bf16)(v > 0.0f ? v : 0.0f);
}

// Fused BN + ReLU + max over K neighbors: (N, K, Nc) bf16 -> (N, Nc) fp32.
// ReLU output >= 0, so the max can be initialized at 0.
__global__ void bnmax_kernel(const __bf16* __restrict__ X, const float* __restrict__ scale,
                             const float* __restrict__ shift, float* __restrict__ out, int Nc) {
    size_t gid = (size_t)blockIdx.x * blockDim.x + threadIdx.x;
    if (gid >= (size_t)NPTS * Nc) return;
    int i = (int)(gid / Nc), c = (int)(gid % Nc);
    float sc = scale[c], sh = shift[c];
    float m = 0.0f;
#pragma unroll
    for (int k = 0; k < KNN; k++) {
        float v = (float)X[((size_t)i * KNN + k) * Nc + c] * sc + sh;
        m = v > m ? v : m;
    }
    out[gid] = m;
}

// ---------------------------------------------------------------------------
static void launch_gemm(int Kd, int Nc, const __bf16* A, const __bf16* B,
                        const float* bias, __bf16* C, hipStream_t s) {
    dim3 grid(Nc / 64, MROWS / 128);
    if      (Kd ==   32 && Nc ==   64) gemm_wmma_kernel<  32,   64><<<grid, 256, 0, s>>>(A, B, bias, C);
    else if (Kd ==   64 && Nc ==   64) gemm_wmma_kernel<  64,   64><<<grid, 256, 0, s>>>(A, B, bias, C);
    else if (Kd ==  128 && Nc ==  128) gemm_wmma_kernel< 128,  128><<<grid, 256, 0, s>>>(A, B, bias, C);
    else if (Kd ==  256 && Nc ==  256) gemm_wmma_kernel< 256,  256><<<grid, 256, 0, s>>>(A, B, bias, C);
    else if (Kd ==  512 && Nc == 1024) gemm_wmma_kernel< 512, 1024><<<grid, 256, 0, s>>>(A, B, bias, C);
    else if (Kd == 1024 && Nc == 1024) gemm_wmma_kernel<1024, 1024><<<grid, 256, 0, s>>>(A, B, bias, C);
}

extern "C" void kernel_launch(void* const* d_in, const int* in_sizes, int n_in,
                              void* d_out, int out_size, void* d_ws, size_t ws_size,
                              hipStream_t stream) {
    (void)in_sizes; (void)n_in; (void)out_size; (void)ws_size;
    const float* x = (const float*)d_in[0];

    char* ws = (char*)d_ws;
    size_t off = 0;
    auto alloc = [&](size_t bytes) -> void* {
        void* p = ws + off;
        off += (bytes + 255) & ~(size_t)255;
        return p;
    };

    int*    idx  = (int*)   alloc((size_t)BSIZE * NPTS * KNN * sizeof(int));
    __bf16* T    = (__bf16*)alloc((size_t)MROWS * 512 * sizeof(__bf16));
    __bf16* Wb   = (__bf16*)alloc((size_t)1024 * 1024 * sizeof(__bf16));
    __bf16* G    = (__bf16*)alloc((size_t)MROWS * 1024 * sizeof(__bf16));
    __bf16* Gn   = (__bf16*)alloc((size_t)MROWS * 1024 * sizeof(__bf16));
    float*  hA   = (float*) alloc((size_t)NPTS * 1024 * sizeof(float));
    float*  hB   = (float*) alloc((size_t)NPTS * 1024 * sizeof(float));
    float*  ssum = (float*) alloc(1024 * sizeof(float));
    float*  ssq  = (float*) alloc(1024 * sizeof(float));
    float*  scl  = (float*) alloc(1024 * sizeof(float));
    float*  shf  = (float*) alloc(1024 * sizeof(float));

    // KNN graph for all batches (reused by every layer).
    knn_kernel<<<(BSIZE * NPTS + 127) / 128, 128, 0, stream>>>(x, idx);

    const int CIN[4]  = {3, 64, 128, 256};
    const int COUT[4] = {64, 128, 256, 1024};
    float* hbufs[2] = {hA, hB};

    for (int b = 0; b < BSIZE; b++) {
        const float* hin = x + (size_t)b * NPTS * DDIM;
        const int* idx_b = idx + (size_t)b * NPTS * KNN;

        for (int l = 0; l < 4; l++) {
            int cin = CIN[l], cout = COUT[l];
            int Kd1 = (2 * cin + 31) & ~31;   // padded K for first GEMM

            const float* w1  = (const float*)d_in[1 + 8 * l + 0];
            const float* b1  = (const float*)d_in[1 + 8 * l + 1];
            const float* g1  = (const float*)d_in[1 + 8 * l + 2];
            const float* be1 = (const float*)d_in[1 + 8 * l + 3];
            const float* w2  = (const float*)d_in[1 + 8 * l + 4];
            const float* b2  = (const float*)d_in[1 + 8 * l + 5];
            const float* g2  = (const float*)d_in[1 + 8 * l + 6];
            const float* be2 = (const float*)d_in[1 + 8 * l + 7];

            // ---- mlp1: t @ w1 + b1, BN, ReLU ----
            convert_w_kernel<<<(Kd1 * cout + 255) / 256, 256, 0, stream>>>(w1, Wb, 2 * cin, Kd1, cout);
            {
                size_t tot = (size_t)MROWS * Kd1;
                build_t_kernel<<<(unsigned)((tot + 255) / 256), 256, 0, stream>>>(hin, idx_b, T, cin, Kd1);
            }
            launch_gemm(Kd1, cout, T, Wb, b1, G, stream);
            colstats_kernel<<<cout, 256, 0, stream>>>(G, MROWS, cout, ssum, ssq);
            bnfinalize_kernel<<<1, cout, 0, stream>>>(ssum, ssq, g1, be1, MROWS, scl, shf);
            {
                size_t tot = (size_t)MROWS * cout;
                bnrelu_kernel<<<(unsigned)((tot + 255) / 256), 256, 0, stream>>>(G, scl, shf, Gn, MROWS, cout);
            }

            // ---- mlp2: m @ w2 + b2, BN, ReLU fused into max-over-K ----
            convert_w_kernel<<<(cout * cout + 255) / 256, 256, 0, stream>>>(w2, Wb, cout, cout, cout);
            launch_gemm(cout, cout, Gn, Wb, b2, G, stream);
            colstats_kernel<<<cout, 256, 0, stream>>>(G, MROWS, cout, ssum, ssq);
            bnfinalize_kernel<<<1, cout, 0, stream>>>(ssum, ssq, g2, be2, MROWS, scl, shf);

            float* hout = (l == 3) ? ((float*)d_out + (size_t)b * NPTS * 1024) : hbufs[l & 1];
            {
                size_t tot = (size_t)NPTS * cout;
                bnmax_kernel<<<(unsigned)((tot + 255) / 256), 256, 0, stream>>>(G, scl, shf, hout, cout);
            }
            hin = hout;
        }
    }
}
